// VectorQuantizer_18279380812065
// MI455X (gfx1250) — matrix operand order repaired and checked by reading, not compiled, not run on hardware
//
#include <hip/hip_runtime.h>
#include <hip/hip_bf16.h>

typedef __attribute__((ext_vector_type(16))) _Float16 v16h;
typedef __attribute__((ext_vector_type(8)))  float    v8f;

#define EMB_DIM   256
#define NUM_EMB   4096
#define N_ROWS    65536            // 64*32*32
#define M_TILE    64               // rows per block (4 waves x 16)
#define NT        32               // codebook rows per LDS tile
#define LDS_STR   264              // half-elements per row (256 + 8 pad)
#define TILE_BYTES (NT * LDS_STR * 2)   // 16896

// ---------------------------------------------------------------- kernel 1
// codebook f32 -> f16, per-code squared norm (f32 exact), zero loss accum
__global__ __launch_bounds__(256) void vq_prep_kernel(
    const float* __restrict__ cb, _Float16* __restrict__ cb16,
    float* __restrict__ cnorm, float* __restrict__ lacc)
{
    const int k = blockIdx.x;        // code id
    const int d = threadIdx.x;       // dim
    __shared__ float red[256];
    float v = cb[(size_t)k * EMB_DIM + d];
    cb16[(size_t)k * EMB_DIM + d] = (_Float16)v;
    red[d] = v * v;
    __syncthreads();
    #pragma unroll
    for (int s = 128; s > 0; s >>= 1) {
        if (d < s) red[d] += red[d + s];
        __syncthreads();
    }
    if (d == 0) cnorm[k] = red[0];
    if (k == 0 && d == 0) *lacc = 0.0f;
}

// ---------------------------------------------------------------- helpers
// Async global -> LDS copy of one NT x 256 f16 tile (16 KB), 8 x b128/thread.
// ASYNCcnt-tracked; data never touches VGPRs.
__device__ __forceinline__ void async_tile_load(_Float16* dstbuf,
                                                const _Float16* src, int tid)
{
    #pragma unroll
    for (int j = 0; j < 8; ++j) {
        const int q  = tid + 128 * j;     // 1024 x 16B chunks
        const int r  = q >> 5;            // row 0..31
        const int c8 = q & 31;            // 16B chunk within row
        const unsigned lds_off =
            (unsigned)(uintptr_t)(dstbuf + r * LDS_STR + c8 * 8);
        const unsigned long long ga =
            (unsigned long long)(uintptr_t)(src + (size_t)r * EMB_DIM + c8 * 8);
        asm volatile("global_load_async_to_lds_b128 %0, %1, off"
                     :: "v"(lds_off), "v"(ga) : "memory");
    }
}

__device__ __forceinline__ void wait_async0()
{
    asm volatile("s_wait_asynccnt 0" ::: "memory");
}

// ---------------------------------------------------------------- kernel 2
// WMMA argmin: each wave computes 16 rows x all 4096 codes, K=256.
// Codebook tiles double-buffered in LDS via async-to-LDS DMA.
__global__ __launch_bounds__(128) void vq_argmin_kernel(
    const float* __restrict__ z, const _Float16* __restrict__ cb16,
    const float* __restrict__ cnorm, int* __restrict__ out_idx)
{
    // One 33KB arena: z-staging (64x264 halfs) overlaps the two codebook
    // tile buffers (2 x 32x264 halfs) -- z is dead once A is in VGPRs.
    __shared__ __align__(16) unsigned char smem[2 * TILE_BYTES];
    _Float16* zs = (_Float16*)smem;

    const int tid  = threadIdx.x;
    const int wave = tid >> 5;
    const int lane = tid & 31;
    const int mrel = lane & 15;      // row (A) / col (B) within 16
    const int grp  = lane >> 4;      // lane half per ISA VGPR layouts
    const int row_base = blockIdx.x * M_TILE;

    // ---- stage z tile to LDS as f16 (float4 global loads, coalesced)
    for (int i = tid; i < M_TILE * (EMB_DIM / 4); i += 128) {
        const int r  = i / (EMB_DIM / 4);
        const int c4 = i % (EMB_DIM / 4);
        const float4 v = ((const float4*)(z + (size_t)(row_base + r) * EMB_DIM))[c4];
        _Float16* dst = &zs[r * LDS_STR + c4 * 4];
        dst[0] = (_Float16)v.x; dst[1] = (_Float16)v.y;
        dst[2] = (_Float16)v.z; dst[3] = (_Float16)v.w;
    }
    __syncthreads();

    // ---- load all A fragments (K=256 -> 8 chunks of 32) into VGPRs
    // 16-bit A 16x32 layout: lane m=l%16; elem i -> K = i+16*grp (matches B)
    v16h A[8];
    {
        const _Float16* src = &zs[(wave * 16 + mrel) * LDS_STR];
        #pragma unroll
        for (int c = 0; c < 8; ++c) {
            #pragma unroll
            for (int i = 0; i < 16; ++i) {
                const int k = 32 * c + i + 16 * grp;
                A[c][i] = src[k];
            }
        }
    }
    __syncthreads();   // z arena now dead; safe to overwrite with tile 0

    float bestv[8];
    int   besti[8];
    #pragma unroll
    for (int r = 0; r < 8; ++r) { bestv[r] = 3.4e38f; besti[r] = 0; }

    // prime the pipeline: tile 0 -> buffer 0
    async_tile_load((_Float16*)&smem[0], cb16, tid);

    const int NUM_TILES = NUM_EMB / NT;   // 128
    for (int it = 0; it < NUM_TILES; ++it) {
        const int nb = it * NT;
        // code norms straight from global (L2-resident), issued early
        const float cn0 = cnorm[nb + mrel];
        const float cn1 = cnorm[nb + 16 + mrel];

        wait_async0();        // this wave's copy of tile `it` has landed
        __syncthreads();      // all waves landed their slices; prev buffer free
        if (it + 1 < NUM_TILES)
            async_tile_load((_Float16*)&smem[((it + 1) & 1) * TILE_BYTES],
                            cb16 + (size_t)(nb + NT) * EMB_DIM, tid);

        const _Float16* cbuf = (const _Float16*)&smem[(it & 1) * TILE_BYTES];
        #pragma unroll
        for (int t = 0; t < NT / 16; ++t) {
            v8f acc = {};
            #pragma unroll
            for (int c = 0; c < 8; ++c) {
                // 16-bit B 32x16 layout (contiguous k, not ISA order): lane col n=l%16; elem i -> K = i+16*grp
                v16h B;
                const _Float16* src =
                    &cbuf[(t * 16 + mrel) * LDS_STR + 32 * c + 16 * grp];
                #pragma unroll
                for (int i = 0; i < 16; ++i) B[i] = src[i];
                acc = __builtin_amdgcn_wmma_f32_16x16x32_f16(
                    false, A[c], false, B, (short)0, acc, false, false);
            }
            const int   nglob = nb + t * 16 + mrel;
            const float cn    = t ? cn1 : cn0;
            // C layout: VGPR r -> M = r + 8*grp, N = lane%16
            #pragma unroll
            for (int r = 0; r < 8; ++r) {
                const float score = cn - 2.0f * acc[r];
                if (score < bestv[r]) { bestv[r] = score; besti[r] = nglob; }
            }
        }
    }

    // ---- reduce (min,idx) across the 16-lane groups (first-index tie-break)
    #pragma unroll
    for (int m = 8; m >= 1; m >>= 1) {
        #pragma unroll
        for (int r = 0; r < 8; ++r) {
            const float ov = __shfl_xor(bestv[r], m, 16);
            const int   oi = __shfl_xor(besti[r], m, 16);
            if (ov < bestv[r] || (ov == bestv[r] && oi < besti[r])) {
                bestv[r] = ov; besti[r] = oi;
            }
        }
    }
    if (mrel == 0) {
        #pragma unroll
        for (int r = 0; r < 8; ++r)
            out_idx[row_base + wave * 16 + r + 8 * grp] = besti[r];
    }
}

// ---------------------------------------------------------------- kernel 3
// gather z_q from the f32 codebook, write output, accumulate sum((z_q-z)^2)
__global__ __launch_bounds__(256) void vq_gather_kernel(
    const float* __restrict__ z, const float* __restrict__ cb,
    const int* __restrict__ idx, float* __restrict__ out,
    float* __restrict__ lacc)
{
    const int gid = blockIdx.x * 256 + threadIdx.x;   // one float4 each
    const int row = gid >> 6;
    const int c4  = gid & 63;
    const int k   = idx[row];
    const float4 e  = ((const float4*)(cb + (size_t)k   * EMB_DIM))[c4];
    const float4 zv = ((const float4*)(z  + (size_t)row * EMB_DIM))[c4];
    ((float4*)(out + (size_t)row * EMB_DIM))[c4] = e;
    const float dx = e.x - zv.x, dy = e.y - zv.y,
                dz = e.z - zv.z, dw = e.w - zv.w;
    float s = dx * dx + dy * dy + dz * dz + dw * dw;
    #pragma unroll
    for (int off = 16; off > 0; off >>= 1) s += __shfl_down(s, off, 32);
    __shared__ float wsum[8];
    const int w = threadIdx.x >> 5, ln = threadIdx.x & 31;
    if (ln == 0) wsum[w] = s;
    __syncthreads();
    if (threadIdx.x == 0) {
        float t = 0.0f;
        #pragma unroll
        for (int i = 0; i < 8; ++i) t += wsum[i];
        atomicAdd(lacc, t);
    }
}

// ---------------------------------------------------------------- kernel 4
__global__ void vq_finalize_kernel(const float* __restrict__ lacc,
                                   float* __restrict__ loss_out)
{
    // loss = mean + COMMIT*mean = 1.25 * sumsq / (N*D)
    *loss_out = 1.25f * (*lacc) / ((float)N_ROWS * (float)EMB_DIM);
}

// ----------------------------------------------------------------
extern "C" void kernel_launch(void* const* d_in, const int* in_sizes, int n_in,
                              void* d_out, int out_size, void* d_ws, size_t ws_size,
                              hipStream_t stream) {
    const float* z  = (const float*)d_in[0];      // [65536, 256]
    const float* cb = (const float*)d_in[1];      // [4096, 256]
    float* out = (float*)d_out;                   // z_q (16777216) + loss (1)

    char* ws = (char*)d_ws;
    _Float16* cb16  = (_Float16*)ws;                                  // 2 MB
    float*    cnorm = (float*)(ws + 2u * 1024u * 1024u);              // 16 KB
    int*      idxb  = (int*)(ws + 2u * 1024u * 1024u + 16384u);       // 256 KB
    float*    lacc  = (float*)(ws + 2u * 1024u * 1024u + 16384u + 262144u);

    vq_prep_kernel<<<NUM_EMB, 256, 0, stream>>>(cb, cb16, cnorm, lacc);
    vq_argmin_kernel<<<N_ROWS / M_TILE, 128, 0, stream>>>(z, cb16, cnorm, idxb);
    vq_gather_kernel<<<(N_ROWS * (EMB_DIM / 4)) / 256, 256, 0, stream>>>(
        z, cb, idxb, out, lacc);
    vq_finalize_kernel<<<1, 1, 0, stream>>>(lacc, out + (size_t)(out_size - 1));
}
